// DiffusionEdgePredictor_214748365349
// MI455X (gfx1250) — matrix operand order repaired and checked by reading, not compiled, not run on hardware
//
#include <hip/hip_runtime.h>
#include <hip/hip_bf16.h>

// Problem constants (from reference)
#define BB 8
#define NN 128
#define FFe 128
#define TFT 64
#define TMT 256
#define DFD 32
#define DMD 128
#define EE 64
#define HH 128
#define TET 64

typedef __attribute__((ext_vector_type(16))) __bf16 v16bf;
typedef __attribute__((ext_vector_type(8)))  float  v8f;

__device__ __forceinline__ unsigned short f2bfbits(float f) {
  union { __bf16 b; unsigned short u; } v; v.b = (__bf16)f; return v.u;
}
__device__ __forceinline__ __bf16 bits2bf(unsigned short s) {
  union { unsigned short s; __bf16 b; } v; v.s = s; return v.b;
}
__device__ __forceinline__ float bfbits2f(unsigned short s) {
  union { unsigned u; float f; } v; v.u = ((unsigned)s) << 16; return v.f;
}

// ---- Vectorized WMMA fragment loaders (16x16x32 bf16, wave32; k order chosen here, same for A and B) ----
// A 16x32 (MxK): lane L holds row M=L%16; element e -> k = (L/16)*16 + e
//   => one contiguous 16-element run starting at kb (loaded as kb and kb+8)
// B 32x16 (KxN): lane L holds col N=L%16; element e -> k = (L/16)*16 + e
//   => one contiguous 16-element (32B) run starting at kb
__device__ __forceinline__ v16bf frag_u16_runs(const unsigned short* run0,
                                               const unsigned short* run1) {
  union { uint4 q[2]; v16bf v; } U;
  U.q[0] = *(const uint4*)run0;   // 8 bf16
  U.q[1] = *(const uint4*)run1;   // 8 bf16
  return U.v;
}
__device__ __forceinline__ v16bf frag_f32_runs(const float* run0, const float* run1) {
  float4 a0 = *(const float4*)(run0);
  float4 a1 = *(const float4*)(run0 + 4);
  float4 b0 = *(const float4*)(run1);
  float4 b1 = *(const float4*)(run1 + 4);
  v16bf v;
  v[0] = (__bf16)a0.x; v[1] = (__bf16)a0.y; v[2] = (__bf16)a0.z; v[3] = (__bf16)a0.w;
  v[4] = (__bf16)a1.x; v[5] = (__bf16)a1.y; v[6] = (__bf16)a1.z; v[7] = (__bf16)a1.w;
  v[8] = (__bf16)b0.x; v[9] = (__bf16)b0.y; v[10] = (__bf16)b0.z; v[11] = (__bf16)b0.w;
  v[12] = (__bf16)b1.x; v[13] = (__bf16)b1.y; v[14] = (__bf16)b1.z; v[15] = (__bf16)b1.w;
  return v;
}
// C 16x16 f32: VGPR i holds row M = i (+8 for lanes16-31), col N = lane%16
template <typename Fn>
__device__ __forceinline__ void store_c16x16(int lane, const v8f& c, Fn wr) {
  const int col = lane & 15;
  const int rb  = (lane >> 4) << 3;
#pragma unroll
  for (int i = 0; i < 8; ++i) wr(i + rb, col, c[i]);
}

// ================= Kernel 1: kv(bf16) = feat_music @ music_W + music_b =================
// grid 16 blocks (b, m-half), 256 threads (8 waves). M=128 N=128 K=128.
__global__ void __launch_bounds__(256) kv_kernel(
    const float* __restrict__ feat_music, const float* __restrict__ music_W,
    const float* __restrict__ music_b, unsigned short* __restrict__ kvbf) {
  __shared__ __align__(16) unsigned short mwT_s[HH * DMD];  // 32KB, transposed [h][d]
  const int tid = threadIdx.x, lane = tid & 31, wave = tid >> 5;
  const int b = blockIdx.x >> 1, m0 = (blockIdx.x & 1) * 128;
  // transpose music_W (d,h) -> (h,d) in bf16
  for (int idx = tid; idx < HH * DMD; idx += 256) {
    int h = idx >> 7, d = idx & 127;
    mwT_s[h * DMD + d] = f2bfbits(music_W[d * HH + h]);
  }
  __syncthreads();
  const float* am = feat_music + (size_t)(b * TMT + m0) * DMD;
  unsigned short* outp = kvbf + (size_t)(b * TMT + m0) * HH;
  const int kh = (lane >> 4) << 3;   // unused; A uses kb like B
  const int kb = (lane >> 4) << 4;   // B run
  const int mrow = lane & 15, ncol = lane & 15;
#pragma unroll 1
  for (int s = 0; s < 8; ++s) {
    int tt = wave * 8 + s;
    int mi = tt >> 3, ni = tt & 7;
    const float* arow = am + (size_t)(mi * 16 + mrow) * DMD;
    const unsigned short* brow = mwT_s + (ni * 16 + ncol) * DMD + kb;
    v8f c = {};
#pragma unroll
    for (int kk = 0; kk < 4; ++kk) {
      int k0 = kk * 32;
      v16bf a = frag_f32_runs(arow + k0 + kb, arow + k0 + kb + 8);
      v16bf bm = frag_u16_runs(brow + k0, brow + k0 + 8);
      c = __builtin_amdgcn_wmma_f32_16x16x32_bf16(false, a, false, bm, (short)0, c, false, false);
    }
    store_c16x16(lane, c, [&](int r, int cc, float v) {
      int row = mi * 16 + r, col = ni * 16 + cc;
      outp[(size_t)row * HH + col] = f2bfbits(v + music_b[col]);
    });
  }
}

// ========== Kernel 2: fused attention per (b,f): scores+softmax+stats+cond_t ==========
// grid 512 blocks = B*Tf, 256 threads (8 waves). n processed in 4 chunks of 32.
// kv is staged to LDS once per block (64KB) via CDNA5 async-DMA (no VGPR round-trip),
// overlapped with the fnirs_W transpose; drained with s_wait_asynccnt.
__global__ void __launch_bounds__(256) attn_kernel(
    const float* __restrict__ feat_fnirs, const float* __restrict__ fnirs_W,
    const float* __restrict__ fnirs_b, const unsigned short* __restrict__ kvbf,
    const int* __restrict__ t, float* __restrict__ cond_t,
    float* __restrict__ ent_part, float* __restrict__ d_out) {
  __shared__ __align__(16) unsigned short kv_s[TMT * HH];   // 64KB  kv[b] bf16
  __shared__ __align__(16) unsigned short wfT_s[HH * DFD];  // 8KB   fnirs_W^T [h][d]
  __shared__ __align__(16) unsigned short q_s[32 * HH];     // 8KB   bf16 Q chunk
  __shared__ __align__(16) float s_s[32 * TMT];             // 32KB  scores/attn chunk
  __shared__ float aw_s[TMT];
  __shared__ float red_s[256];
  __shared__ float rmax_s[32], rsum_s[32];
  __shared__ float ent_s[256];

  const int tid = threadIdx.x, lane = tid & 31, wave = tid >> 5;
  const int b = blockIdx.x >> 6, f = blockIdx.x & 63;
  int tb = t[b]; tb = tb < 0 ? 0 : (tb > TFT - 1 ? TFT - 1 : tb);
  const bool isT = (f == tb);
  const float scale = 0.08838834764831845f;  // 1/sqrt(H)
  const unsigned short* kvb = kvbf + (size_t)b * TMT * HH;
  const int kh = (lane >> 4) << 3;
  const int kb = (lane >> 4) << 4;
  const int mrow = lane & 15, ncol = lane & 15;
  float ent_local = 0.0f;
  aw_s[tid] = 0.0f;

  // ---- stage kv[b] into LDS with async DMA: 256 lanes x 16 x b128 = 64KB ----
  {
    const unsigned long long gbase = (unsigned long long)kvb;
    const unsigned ldsbase = (unsigned)(size_t)kv_s;  // LDS aperture: low 32 bits
#pragma unroll 1
    for (int it = 0; it < 16; ++it) {
      unsigned off = (unsigned)(it * 256 + tid) * 16u;  // byte offset
      asm volatile("global_load_async_to_lds_b128 %0, %1, %2"
                   :
                   : "v"(ldsbase + off), "v"(off), "s"(gbase)
                   : "memory");
    }
  }
  // transpose fnirs_W (d,h) -> (h,d) bf16 (normal loads, overlaps with DMA)
  for (int idx = tid; idx < HH * DFD; idx += 256) {
    int h = idx >> 5, d = idx & 31;
    wfT_s[h * DFD + d] = f2bfbits(fnirs_W[d * HH + h]);
  }
  asm volatile("s_wait_asynccnt 0x0" ::: "memory");
  __syncthreads();

#pragma unroll 1
  for (int chunk = 0; chunk < 4; ++chunk) {
    const int n0 = chunk * 32;
    // prefetch next chunk's fnirs rows toward the WGP while we compute
    if (chunk < 3 && tid < 32) {
      const float* nxt = feat_fnirs +
          ((size_t)(b * NN + n0 + 32 + tid) * TFT + f) * DFD;
      __builtin_prefetch(nxt, 0, 3);
    }
    // ---- Q chunk (32 x 128) = Ff(32x32) @ Wf(32x128) + b ; one K-step (Df=32) ----
#pragma unroll
    for (int s = 0; s < 2; ++s) {
      int tt = wave * 2 + s;
      int mi = tt >> 3, ni = tt & 7;
      const float* fr = feat_fnirs +
          ((size_t)(b * NN + n0 + mi * 16 + mrow) * TFT + f) * DFD;
      v16bf a = frag_f32_runs(fr + kb, fr + kb + 8);
      const unsigned short* bp = wfT_s + (ni * 16 + ncol) * DFD + kb;
      v16bf bm = frag_u16_runs(bp, bp + 8);
      v8f c = {};
      c = __builtin_amdgcn_wmma_f32_16x16x32_bf16(false, a, false, bm, (short)0, c, false, false);
      store_c16x16(lane, c, [&](int r, int cc, float v) {
        int row = mi * 16 + r, col = ni * 16 + cc;
        q_s[row * HH + col] = f2bfbits(v + fnirs_b[col]);
      });
    }
    __syncthreads();
    // ---- scores (32 x 256) = Q @ kv^T, K=128; kv served from LDS ----
#pragma unroll 1
    for (int s = 0; s < 4; ++s) {
      int tt = wave * 4 + s;
      int mi = tt >> 4, mj = tt & 15;
      const unsigned short* arow = q_s + (mi * 16 + mrow) * HH;
      const unsigned short* brow = kv_s + (mj * 16 + ncol) * HH + kb;
      v8f c = {};
#pragma unroll
      for (int kk = 0; kk < 4; ++kk) {
        int k0 = kk * 32;
        v16bf a = frag_u16_runs(arow + k0 + kb, arow + k0 + kb + 8);
        v16bf bm = frag_u16_runs(brow + k0, brow + k0 + 8);
        c = __builtin_amdgcn_wmma_f32_16x16x32_bf16(false, a, false, bm, (short)0, c, false, false);
      }
      store_c16x16(lane, c, [&](int r, int cc, float v) {
        s_s[(mi * 16 + r) * TMT + mj * 16 + cc] = v * scale;
      });
    }
    __syncthreads();
    // ---- softmax: 32 rows x 256 cols, 8 threads per row ----
    {
      const int row = tid >> 3, part = tid & 7;
      float* srow = s_s + row * TMT + part * 32;
      float mx = -3.0e38f;
#pragma unroll 4
      for (int j = 0; j < 32; ++j) mx = fmaxf(mx, srow[j]);
      red_s[tid] = mx;
      __syncthreads();
      if (part == 0) {
        float m2 = red_s[tid];
        for (int j = 1; j < 8; ++j) m2 = fmaxf(m2, red_s[tid + j]);
        rmax_s[row] = m2;
      }
      __syncthreads();
      float rm = rmax_s[row], ps = 0.0f;
#pragma unroll 4
      for (int j = 0; j < 32; ++j) { float e = __expf(srow[j] - rm); srow[j] = e; ps += e; }
      red_s[tid] = ps;
      __syncthreads();
      if (part == 0) {
        float s2 = 0.0f;
        for (int j = 0; j < 8; ++j) s2 += red_s[tid + j];
        rsum_s[row] = s2;
      }
      __syncthreads();
      float inv = 1.0f / rsum_s[row];
#pragma unroll 4
      for (int j = 0; j < 32; ++j) {
        float a = srow[j] * inv; srow[j] = a;
        ent_local += a * __logf(a + 1e-8f);
      }
    }
    __syncthreads();
    // ---- attn_weights: reduce over this n-chunk (block owns full n-reduction) ----
    {
      float acc = 0.0f;
#pragma unroll 4
      for (int r = 0; r < 32; ++r) acc += s_s[r * TMT + tid];
      aw_s[tid] += acc;
    }
    // ---- cond_t rows: only the f == t[b] block computes ctx (64x cheaper) ----
    if (isT) {
#pragma unroll 1
      for (int s = 0; s < 2; ++s) {
        int tt = wave * 2 + s;
        int mi = tt >> 3, ni = tt & 7;
        const float* arow = s_s + (mi * 16 + mrow) * TMT;
        v8f c = {};
#pragma unroll 1
        for (int kk = 0; kk < 8; ++kk) {
          int k0 = kk * 32;
          v16bf a = frag_f32_runs(arow + k0 + kb, arow + k0 + kb + 8);
          v16bf bm;
#pragma unroll
          for (int e = 0; e < 16; ++e)  // B[k=m][n=h] = kv[m][h], strided (rare path, LDS)
            bm[e] = bits2bf(kv_s[(k0 + kb + e) * HH + ni * 16 + ncol]);
          c = __builtin_amdgcn_wmma_f32_16x16x32_bf16(false, a, false, bm, (short)0, c, false, false);
        }
        store_c16x16(lane, c, [&](int r, int cc, float v) {
          int rowl = mi * 16 + r, col = ni * 16 + cc;
          float qf = bfbits2f(q_s[rowl * HH + col]);
          cond_t[(size_t)(b * NN + n0 + rowl) * HH + col] = v + qf;
        });
      }
    }
    __syncthreads();
  }
  // attn_weights out (mean over N)
  d_out[BB * NN * NN + (b * TFT + f) * TMT + tid] = aw_s[tid] * (1.0f / (float)NN);
  // deterministic entropy: per-block partial to workspace
  ent_s[tid] = ent_local;
  __syncthreads();
  for (int st = 128; st > 0; st >>= 1) {
    if (tid < st) ent_s[tid] += ent_s[tid + st];
    __syncthreads();
  }
  if (tid == 0) ent_part[blockIdx.x] = ent_s[0];
}

// ========== Kernel 3: deterministic loss reduction over 512 block partials ==========
__global__ void __launch_bounds__(256) loss_kernel(const float* __restrict__ ent_part,
                                                   float* __restrict__ loss) {
  __shared__ float red[256];
  int tid = threadIdx.x;
  red[tid] = ent_part[tid] + ent_part[tid + 256];
  __syncthreads();
  for (int st = 128; st > 0; st >>= 1) {
    if (tid < st) red[tid] += red[tid + st];
    __syncthreads();
  }
  if (tid == 0) *loss = -red[0] * (1.0f / (float)(BB * NN * TFT));
}

// ========== Kernel 4: h = node_feats @ node_W + node_b ==========
__global__ void h_kernel(const float* __restrict__ node_feats, const float* __restrict__ node_W,
                         const float* __restrict__ node_b, float* __restrict__ hbuf) {
  int idx = blockIdx.x * blockDim.x + threadIdx.x;
  if (idx >= BB * NN * EE) return;
  int e = idx % EE, bi = idx / EE;
  const float* nf = node_feats + (size_t)bi * FFe;
  float acc = node_b[e];
  for (int k = 0; k < FFe; ++k) acc += nf[k] * node_W[k * EE + e];
  hbuf[idx] = acc;
}

// ========== Kernel 5: tc[b,h] = time_emb(t_b) @ W1_time + mlp_b1 ==========
__global__ void tc_kernel(const int* __restrict__ t, const float* __restrict__ mlp_W1,
                          const float* __restrict__ mlp_b1, float* __restrict__ tc) {
  int idx = blockIdx.x * blockDim.x + threadIdx.x;
  if (idx >= BB * HH) return;
  int b = idx / HH, hh = idx % HH;
  int tb = t[b]; tb = tb < 0 ? 0 : (tb > TFT - 1 ? TFT - 1 : tb);
  float tv = (float)tb;
  const float lco = 0.2971077539435f;  // ln(10000)/31
  float acc = mlp_b1[hh];
  const float* wt = mlp_W1 + (2 * EE + 2 + HH) * HH;
  for (int k = 0; k < TET; ++k) {
    int kk = (k < 32) ? k : (k - 32);
    float fr = tv * __expf(-(float)kk * lco);
    float te = (k < 32) ? __sinf(fr) : __cosf(fr);
    acc += te * wt[k * HH + hh];
  }
  tc[idx] = acc;
}

// ========== Kernel 6: hi = h @ W1_hi ==========
__global__ void hi_kernel(const float* __restrict__ hbuf, const float* __restrict__ mlp_W1,
                          float* __restrict__ hi) {
  int idx = blockIdx.x * blockDim.x + threadIdx.x;
  if (idx >= BB * NN * HH) return;
  int hh = idx % HH, bi = idx / HH;
  const float* hr = hbuf + (size_t)bi * EE;
  float acc = 0.0f;
  for (int e = 0; e < EE; ++e) acc += hr[e] * mlp_W1[e * HH + hh];
  hi[idx] = acc;
}

// ========== Kernel 7: hjcT[b,h,j] = (h@W1_hj + cond_t@W1_cond)[j,h] + tc[b,h] ==========
__global__ void hjc_kernel(const float* __restrict__ hbuf, const float* __restrict__ cond_t,
                           const float* __restrict__ tc, const float* __restrict__ mlp_W1,
                           float* __restrict__ hjcT) {
  int idx = blockIdx.x * blockDim.x + threadIdx.x;
  if (idx >= BB * HH * NN) return;
  int j = idx % NN;
  int hh = (idx / NN) % HH;
  int b = idx / (NN * HH);
  const float* hr = hbuf + (size_t)(b * NN + j) * EE;
  const float* cr = cond_t + (size_t)(b * NN + j) * HH;
  float acc = tc[b * HH + hh];
  const float* whj = mlp_W1 + EE * HH;
  for (int e = 0; e < EE; ++e) acc += hr[e] * whj[e * HH + hh];
  const float* wc = mlp_W1 + (2 * EE + 2) * HH;
  for (int c = 0; c < HH; ++c) acc += cr[c] * wc[c * HH + hh];
  hjcT[idx] = acc;
}

// ========== Kernel 8: edge MLP -> noise_pred ==========
// grid B*N blocks (one (b,i) row), 128 threads (one j each). Coalesced in j via hjcT.
__global__ void __launch_bounds__(128) edge_kernel(
    const float* __restrict__ noisy_adj, const float* __restrict__ g_pre,
    const float* __restrict__ hi, const float* __restrict__ hjcT,
    const float* __restrict__ mlp_W1, const float* __restrict__ mlp_W2,
    const float* __restrict__ mlp_b2, float* __restrict__ outp) {
  __shared__ float hi_s[HH], wadj_s[HH], wgp_s[HH], w2_s[HH];
  const int bi = blockIdx.x;
  const int j = threadIdx.x;
  const int b = bi >> 7;
  hi_s[j]   = hi[(size_t)bi * HH + j];
  wadj_s[j] = mlp_W1[(2 * EE) * HH + j];
  wgp_s[j]  = mlp_W1[(2 * EE + 1) * HH + j];
  w2_s[j]   = mlp_W2[j];
  __syncthreads();
  const float na = noisy_adj[(size_t)bi * NN + j];
  const float gp = g_pre[(size_t)bi * NN + j];
  const float* hj = hjcT + (size_t)b * HH * NN;
  float acc = 0.0f;
#pragma unroll 4
  for (int hh = 0; hh < HH; ++hh) {
    float v = hi_s[hh] + hj[hh * NN + j] + na * wadj_s[hh] + gp * wgp_s[hh];
    acc += fmaxf(v, 0.0f) * w2_s[hh];
  }
  outp[(size_t)bi * NN + j] = acc + mlp_b2[0];
}

extern "C" void kernel_launch(void* const* d_in, const int* in_sizes, int n_in,
                              void* d_out, int out_size, void* d_ws, size_t ws_size,
                              hipStream_t stream) {
  const float* noisy_adj  = (const float*)d_in[0];
  const float* node_feats = (const float*)d_in[1];
  const float* feat_fnirs = (const float*)d_in[2];
  const float* feat_music = (const float*)d_in[3];
  const float* g_pre      = (const float*)d_in[4];
  const float* node_W     = (const float*)d_in[5];
  const float* node_b     = (const float*)d_in[6];
  const float* fnirs_W    = (const float*)d_in[7];
  const float* fnirs_b    = (const float*)d_in[8];
  const float* music_W    = (const float*)d_in[9];
  const float* music_b    = (const float*)d_in[10];
  const float* mlp_W1     = (const float*)d_in[11];
  const float* mlp_b1     = (const float*)d_in[12];
  const float* mlp_W2     = (const float*)d_in[13];
  const float* mlp_b2     = (const float*)d_in[14];
  const int*   t          = (const int*)d_in[15];
  float* outp = (float*)d_out;

  // workspace layout (floats): ~2.4 MB total
  float* ws = (float*)d_ws;
  unsigned short* kvbf = (unsigned short*)ws;  // 8*256*128 u16 = 131072 floats
  float* cond_t  = ws + 131072;      // 8*128*128 = 131072
  float* hbuf    = cond_t + 131072;  // 8*128*64  = 65536
  float* tc      = hbuf + 65536;     // 8*128     = 1024
  float* hi      = tc + 1024;        // 8*128*128 = 131072
  float* hjcT    = hi + 131072;      // 8*128*128 = 131072
  float* entpart = hjcT + 131072;    // 512

  kv_kernel  <<<16,   256, 0, stream>>>(feat_music, music_W, music_b, kvbf);
  attn_kernel<<<512,  256, 0, stream>>>(feat_fnirs, fnirs_W, fnirs_b, kvbf, t, cond_t,
                                        entpart, outp);
  loss_kernel<<<1,    256, 0, stream>>>(entpart, outp + (BB * NN * NN + BB * TFT * TMT));
  h_kernel   <<<256,  256, 0, stream>>>(node_feats, node_W, node_b, hbuf);
  tc_kernel  <<<4,    256, 0, stream>>>(t, mlp_W1, mlp_b1, tc);
  hi_kernel  <<<512,  256, 0, stream>>>(hbuf, mlp_W1, hi);
  hjc_kernel <<<512,  256, 0, stream>>>(hbuf, cond_t, tc, mlp_W1, hjcT);
  edge_kernel<<<1024, 128, 0, stream>>>(noisy_adj, g_pre, hi, hjcT, mlp_W1, mlp_W2,
                                        mlp_b2, outp);
}
